// TiPBMRec_69630009803246
// MI455X (gfx1250) — matrix operand order repaired and checked by reading, not compiled, not run on hardware
//
#include <hip/hip_runtime.h>
#include <hip/hip_bf16.h>
#include <math.h>

#define Bsz 128
#define Lp 50
#define Hp 64
#define NHp 2
#define HDp 32
#define NLp 2
#define INNERp 256
#define Vp 50000
#define NTOK (Bsz * Lp)

typedef __attribute__((ext_vector_type(16))) __bf16 v16bf;
typedef __attribute__((ext_vector_type(8)))  float  v8f;

// ---------- wave helpers ----------
__device__ __forceinline__ float wave_sum32(float v) {
  #pragma unroll
  for (int o = 1; o < 32; o <<= 1) v += __shfl_xor(v, o, 32);
  return v;
}
__device__ __forceinline__ float half_sum16(float v) {
  #pragma unroll
  for (int o = 1; o < 16; o <<= 1) v += __shfl_xor(v, o, 32);
  return v;
}
__device__ __forceinline__ float half_max16(float v) {
  #pragma unroll
  for (int o = 1; o < 16; o <<= 1) v = fmaxf(v, __shfl_xor(v, o, 32));
  return v;
}
__device__ __forceinline__ uint32_t lds_addr_of(const void* p) {
  // generic shared pointer: low 32 bits carry the LDS byte offset
  return (uint32_t)(uintptr_t)p;
}
// one async b128 per lane: ldsa = LDS byte address, voff = per-lane byte offset, gb = SGPR base
__device__ __forceinline__ void async_b128(uint32_t ldsa, uint32_t voff, uint64_t gb) {
  asm volatile("global_load_async_to_lds_b128 %0, %1, %2"
               :: "v"(ldsa), "v"(voff), "s"(gb) : "memory");
}

// ---------- generic bf16-WMMA tiled GEMM: C[M,N] = act(A[M,K] @ W[K,N] + bias) ----------
// one wave per 16x16 output tile; K multiple of 32, M,N multiples of 16.
// W tiles streamed global->LDS with the async engine, double buffered over K.
__global__ void gemm_wmma(const float* __restrict__ A, const float* __restrict__ W,
                          const float* __restrict__ bias, float* __restrict__ C,
                          int M, int N, int K, int act) {
  __shared__ __align__(16) float As[16][32];
  __shared__ __align__(16) float Ws2[2][32][16];      // 2 KB per buffer, contiguous
  const int m0 = blockIdx.x * 16;
  const int n0 = blockIdx.y * 16;
  const int lane = threadIdx.x;
  const int r = lane & 15, h = lane >> 4;
  const uint32_t wbase0 = lds_addr_of(&Ws2[0][0][0]);
  // per-lane chunk mapping for one W tile: chunk c = i*32+lane, row=c>>2, colq=c&3
  const int c0row[4] = { (0 * 32 + lane) >> 2, (1 * 32 + lane) >> 2,
                         (2 * 32 + lane) >> 2, (3 * 32 + lane) >> 2 };
  const int c0cq  = lane & 3;
  // prologue: issue k0 = 0 W tile into buffer 0
  {
    uint64_t gb = (uint64_t)(uintptr_t)(W + n0);
    #pragma unroll
    for (int i = 0; i < 4; ++i) {
      uint32_t voff = (uint32_t)(((size_t)c0row[i] * N + (size_t)c0cq * 4) * 4);
      uint32_t ldsa = wbase0 + (uint32_t)(c0row[i] * 64 + c0cq * 16);
      async_b128(ldsa, voff, gb);
    }
  }
  v8f acc = {};
  for (int k0 = 0; k0 < K; k0 += 32) {
    const int cur = (k0 >> 5) & 1;
    // stage A tile synchronously (overlaps the in-flight async W fill)
    #pragma unroll
    for (int i = 0; i < 16; ++i) {
      int idx = lane + i * 32;                       // 0..511
      As[idx >> 5][idx & 31] = A[(size_t)(m0 + (idx >> 5)) * K + (k0 + (idx & 31))];
    }
    // issue next W tile into the other buffer, then wait only for the current one
    if (k0 + 32 < K) {
      uint64_t gb = (uint64_t)(uintptr_t)(W + (size_t)(k0 + 32) * N + n0);
      uint32_t dstb = wbase0 + (uint32_t)((cur ^ 1) << 11);   // 2048 B per buffer
      #pragma unroll
      for (int i = 0; i < 4; ++i) {
        uint32_t voff = (uint32_t)(((size_t)c0row[i] * N + (size_t)c0cq * 4) * 4);
        uint32_t ldsa = dstb + (uint32_t)(c0row[i] * 64 + c0cq * 16);
        async_b128(ldsa, voff, gb);
      }
      asm volatile("s_wait_asynccnt 4" ::: "memory");  // newest 4 outstanding -> cur tile done
    } else {
      asm volatile("s_wait_asynccnt 0" ::: "memory");
    }
    __syncthreads();
    v16bf av, bv;
    #pragma unroll
    for (int j = 0; j < 8; ++j) {
      #pragma unroll
      for (int p = 0; p < 2; ++p) {
        int ka = 16 * (j >> 2) + 8 * h + 2 * (j & 3) + p;   // A lane layout
        av[2 * j + p] = (__bf16)As[r][ka];
        int kb = 16 * (j >> 2) + 8 * h + 2 * (j & 3) + p;   // B lane layout
        bv[2 * j + p] = (__bf16)Ws2[cur][kb][r];
      }
    }
    acc = __builtin_amdgcn_wmma_f32_16x16x32_bf16(false, av, false, bv,
                                                  (short)0, acc, false, false);
    __syncthreads();
  }
  #pragma unroll
  for (int v = 0; v < 8; ++v) {
    int row = m0 + v + 8 * h;
    int col = n0 + r;
    float val = acc[v] + bias[col];
    if (act == 1) val = 0.5f * val * (1.0f + erff(val * 0.70710678118654752f));
    C[(size_t)row * N + col] = val;
  }
}

// ---------- fused X@Y^T row-wise (diag - logsumexp), flash-style, bf16 WMMA ----------
// one wave handles 16 rows; streams all N columns in 16-wide tiles; H = 64.
// Y tiles are contiguous 4 KB blocks -> async global->LDS, double buffered.
__global__ void sim_lse_wmma(const float* __restrict__ X, const float* __restrict__ Y,
                             float* __restrict__ pos, int N) {
  __shared__ __align__(16) float Xs[16][Hp];
  __shared__ __align__(16) float Ybuf[2][16][Hp];     // 4 KB per buffer, contiguous
  const int m0 = blockIdx.x * 16;
  const int lane = threadIdx.x;
  const int r = lane & 15, h = lane >> 4;
  const uint32_t ybase0 = lds_addr_of(&Ybuf[0][0][0]);
  const uint32_t laneb = (uint32_t)lane * 16u;

  // prologue: issue Y tile j0 = 0 into buffer 0 (8 x b128, 512 B each step)
  {
    uint64_t gb = (uint64_t)(uintptr_t)Y;
    #pragma unroll
    for (int i = 0; i < 8; ++i)
      async_b128(ybase0 + (uint32_t)(i * 512) + laneb, (uint32_t)(i * 512) + laneb, gb);
  }
  // stage X rows synchronously and build the A operands once
  #pragma unroll
  for (int i = 0; i < 32; ++i) {
    int idx = lane + i * 32;                         // 0..1023
    Xs[idx >> 6][idx & 63] = X[(size_t)(m0 + (idx >> 6)) * Hp + (idx & 63)];
  }
  __syncthreads();
  v16bf a0, a1;
  #pragma unroll
  for (int j = 0; j < 8; ++j) {
    #pragma unroll
    for (int p = 0; p < 2; ++p) {
      int ka = 16 * (j >> 2) + 8 * h + 2 * (j & 3) + p;
      a0[2 * j + p] = (__bf16)Xs[r][ka];
      a1[2 * j + p] = (__bf16)Xs[r][ka + 32];
    }
  }
  float mrun[8], lrun[8], diag[8];
  #pragma unroll
  for (int v = 0; v < 8; ++v) { mrun[v] = -3.0e38f; lrun[v] = 0.f; diag[v] = 0.f; }

  for (int j0 = 0; j0 < N; j0 += 16) {
    const int cur = (j0 >> 4) & 1;
    if (j0 + 16 < N) {
      uint64_t gb = (uint64_t)(uintptr_t)(Y + (size_t)(j0 + 16) * Hp);
      uint32_t dstb = ybase0 + (uint32_t)((cur ^ 1) << 12);   // 4096 B per buffer
      #pragma unroll
      for (int i = 0; i < 8; ++i)
        async_b128(dstb + (uint32_t)(i * 512) + laneb, (uint32_t)(i * 512) + laneb, gb);
      asm volatile("s_wait_asynccnt 8" ::: "memory");  // newest 8 outstanding -> cur tile done
    } else {
      asm volatile("s_wait_asynccnt 0" ::: "memory");
    }
    __syncthreads();
    v16bf b0, b1;
    #pragma unroll
    for (int j = 0; j < 8; ++j) {
      #pragma unroll
      for (int p = 0; p < 2; ++p) {
        int kb = 16 * (j >> 2) + 8 * h + 2 * (j & 3) + p;
        b0[2 * j + p] = (__bf16)Ybuf[cur][r][kb];     // Ybuf[.][col][k] == Y[j0+col][k]
        b1[2 * j + p] = (__bf16)Ybuf[cur][r][kb + 32];
      }
    }
    v8f s = {};
    s = __builtin_amdgcn_wmma_f32_16x16x32_bf16(false, a0, false, b0, (short)0, s, false, false);
    s = __builtin_amdgcn_wmma_f32_16x16x32_bf16(false, a1, false, b1, (short)0, s, false, false);
    #pragma unroll
    for (int v = 0; v < 8; ++v) {
      float val = s[v];                               // TEMP == 1.0
      if (j0 == m0) diag[v] = __shfl(val, v + 24 * h, 32);  // col == row within tile
      float tm = half_max16(val);
      float mn = fmaxf(mrun[v], tm);
      float e  = half_sum16(expf(val - mn));
      lrun[v] = lrun[v] * expf(mrun[v] - mn) + e;
      mrun[v] = mn;
    }
    __syncthreads();
  }
  if (r == 0) {
    #pragma unroll
    for (int v = 0; v < 8; ++v)
      pos[m0 + v + 8 * h] = diag[v] - (mrun[v] + logf(lrun[v]));
  }
}

// ---------- embedding gather (+optional positional add) ----------
__global__ void gather_kernel(const float* __restrict__ table, const int* __restrict__ seq,
                              const float* __restrict__ pos, float* __restrict__ out) {
  int t = blockIdx.x, d = threadIdx.x;               // 6400 blocks x 64
  float v = table[(size_t)seq[t] * Hp + d];
  if (pos) v += pos[(t % Lp) * Hp + d];
  out[(size_t)t * Hp + d] = v;
}

// ---------- layernorm over H=64, one wave per token, optional residual ----------
__global__ void ln_kernel(const float* __restrict__ in, const float* __restrict__ res,
                          const float* __restrict__ g, const float* __restrict__ b,
                          float* __restrict__ out) {
  int t = blockIdx.x, lane = threadIdx.x;
  size_t o = (size_t)t * Hp;
  float x0 = in[o + lane]      + (res ? res[o + lane]      : 0.f);
  float x1 = in[o + lane + 32] + (res ? res[o + lane + 32] : 0.f);
  float mean = wave_sum32(x0 + x1) * (1.f / Hp);
  float d0 = x0 - mean, d1 = x1 - mean;
  float var = wave_sum32(d0 * d0 + d1 * d1) * (1.f / Hp);
  float inv = rsqrtf(var + 1e-12f);
  out[o + lane]      = d0 * inv * g[lane]      + b[lane];
  out[o + lane + 32] = d1 * inv * g[lane + 32] + b[lane + 32];
}

// ---------- attention: one block per (batch, head); thread = query position ----------
__global__ void attn_kernel(const float* __restrict__ q, const float* __restrict__ k,
                            const float* __restrict__ v, const int* __restrict__ seq,
                            float* __restrict__ ctx) {
  int bh = blockIdx.x;
  int b = bh / NHp, hh = bh % NHp;
  int qi = threadIdx.x;
  if (qi >= Lp) return;
  const float scale = 0.17677669529663687f;          // 1/sqrt(32)
  float p[Lp];
  float mx = -1e30f;
  const float* qrow = q + ((size_t)(b * Lp + qi)) * Hp + hh * HDp;
  for (int kk = 0; kk < Lp; ++kk) {
    const float* krow = k + ((size_t)(b * Lp + kk)) * Hp + hh * HDp;
    float d = 0.f;
    #pragma unroll
    for (int t = 0; t < HDp; ++t) d += qrow[t] * krow[t];
    d = d * scale + (seq[b * Lp + kk] > 0 ? 0.f : -10000.f);
    p[kk] = d; mx = fmaxf(mx, d);
  }
  float s = 0.f;
  for (int kk = 0; kk < Lp; ++kk) { p[kk] = expf(p[kk] - mx); s += p[kk]; }
  float inv = 1.f / s;
  float out[HDp];
  #pragma unroll
  for (int t = 0; t < HDp; ++t) out[t] = 0.f;
  for (int kk = 0; kk < Lp; ++kk) {
    const float* vrow = v + ((size_t)(b * Lp + kk)) * Hp + hh * HDp;
    float pw = p[kk] * inv;
    #pragma unroll
    for (int t = 0; t < HDp; ++t) out[t] += pw * vrow[t];
  }
  float* crow = ctx + ((size_t)(b * Lp + qi)) * Hp + hh * HDp;
  #pragma unroll
  for (int t = 0; t < HDp; ++t) crow[t] = out[t];
}

// ---------- L2-normalize rows (eps clamp at 1e-8) ----------
__global__ void unit_kernel(const float* __restrict__ in, float* __restrict__ out) {
  int t = blockIdx.x, lane = threadIdx.x;
  size_t o = (size_t)t * Hp;
  float x0 = in[o + lane], x1 = in[o + lane + 32];
  float n2 = wave_sum32(x0 * x0 + x1 * x1);
  float inv = 1.f / fmaxf(sqrtf(n2), 1e-8f);
  out[o + lane] = x0 * inv;
  out[o + lane + 32] = x1 * inv;
}

// ---------- global mean of time_intervals ----------
__global__ void tmean_kernel(const float* __restrict__ tiv, float* __restrict__ gmean) {
  __shared__ float sh[256];
  float s = 0.f;
  for (int n = threadIdx.x; n < NTOK; n += 256) s += tiv[n];
  sh[threadIdx.x] = s; __syncthreads();
  for (int off = 128; off; off >>= 1) {
    if ((int)threadIdx.x < off) sh[threadIdx.x] += sh[threadIdx.x + off];
    __syncthreads();
  }
  if (threadIdx.x == 0) gmean[0] = sh[0] / (float)NTOK;
}

// ---------- per-batch conformity weight ----------
__global__ void conf_kernel(const float* __restrict__ tiv, const float* __restrict__ adjG,
                            const float* __restrict__ se, const float* __restrict__ gmean,
                            float* __restrict__ conf) {
  int b = blockIdx.x, lane = threadIdx.x;            // one wave per batch
  float s = 0.f;
  for (int l = lane; l < Lp; l += 32) s += tiv[b * Lp + l];
  s = wave_sum32(s);
  float mean = s / (float)Lp;
  float vs = 0.f;
  for (int l = lane; l < Lp; l += 32) { float d = tiv[b * Lp + l] - mean; vs += d * d; }
  vs = wave_sum32(vs);
  float sd = sqrtf(vs / (float)(Lp - 1));
  float simsum = 0.f;
  for (int l = 0; l < Lp; ++l) {
    size_t o = ((size_t)(b * Lp + l)) * Hp;
    float x0 = adjG[o + lane], x1 = adjG[o + lane + 32];
    float n2 = wave_sum32(x0 * x0 + x1 * x1);
    float cl = fmaxf(sqrtf(n2), 1e-8f);
    simsum += n2 / (cl * cl);
  }
  float indiv = (simsum / (float)Lp) * sd;
  float gm = gmean[0];
  float nu0 = 0.f, nu1 = 0.f, du = 0.f, na0 = 0.f, na1 = 0.f, da = 0.f;
  for (int l = 0; l < Lp; ++l) {
    float tv = tiv[b * Lp + l];
    float eu = expf(-tv / mean);
    float ea = expf(-tv / gm);
    size_t o = ((size_t)(b * Lp + l)) * Hp;
    float e0 = se[o + lane], e1 = se[o + lane + 32];
    nu0 += e0 * eu; nu1 += e1 * eu; du += eu;
    na0 += e0 * ea; na1 += e1 * ea; da += ea;
  }
  float l0 = nu0 / du, l1 = nu1 / du;
  float o0 = na0 / da, o1 = na1 / da;
  float dot = wave_sum32(l0 * o0 + l1 * o1);
  float nl  = fmaxf(sqrtf(wave_sum32(l0 * l0 + l1 * l1)), 1e-8f);
  float no  = fmaxf(sqrtf(wave_sum32(o0 * o0 + o1 * o1)), 1e-8f);
  float cosv = dot / (nl * no);
  if (lane == 0) conf[b] = 0.5f * (indiv + cosv);
}

// ---------- weighted loss reductions + KL, single block of 128 ----------
__global__ void loss_kernel(const float* __restrict__ upos, const float* __restrict__ ipos,
                            const float* __restrict__ conf, float* __restrict__ out) {
  __shared__ float sh[128];
  int t = threadIdx.x;
  float su = 0.f, si = 0.f;
  for (int n = t; n < NTOK; n += 128) {
    float w = conf[n / Lp];
    su += w * upos[n];
    si += (1.f - w) * ipos[n];
  }
  sh[t] = su; __syncthreads();
  for (int off = 64; off; off >>= 1) { if (t < off) sh[t] += sh[t + off]; __syncthreads(); }
  float tot_u = sh[0]; __syncthreads();
  sh[t] = si; __syncthreads();
  for (int off = 64; off; off >>= 1) { if (t < off) sh[t] += sh[t + off]; __syncthreads(); }
  float tot_i = sh[0]; __syncthreads();
  float wf = conf[t];
  sh[t] = wf; __syncthreads();
  for (int off = 64; off; off >>= 1) { if (t < off) sh[t] = fmaxf(sh[t], sh[t + off]); __syncthreads(); }
  float mx = sh[0]; __syncthreads();
  sh[t] = expf(wf - mx); __syncthreads();
  for (int off = 64; off; off >>= 1) { if (t < off) sh[t] += sh[t + off]; __syncthreads(); }
  float lse = mx + logf(sh[0]); __syncthreads();
  float qv = 1.f / (float)Bsz;
  sh[t] = qv * (logf(qv) - (wf - lse)); __syncthreads();
  for (int off = 64; off; off >>= 1) { if (t < off) sh[t] += sh[t + off]; __syncthreads(); }
  if (t == 0) {
    out[0] = -tot_u / (float)NTOK;
    out[1] = -tot_i / (float)NTOK;
    out[2] = sh[0] / (float)Bsz;
  }
}

// ---------- concat last-position features -> [B, 3H] ----------
__global__ void combined_kernel(const float* __restrict__ x, const float* __restrict__ adjG,
                                const float* __restrict__ coG, float* __restrict__ comb) {
  int b = blockIdx.x, d = threadIdx.x;               // 192 threads
  size_t last = ((size_t)b * Lp + (Lp - 1)) * Hp;
  float v;
  if (d < Hp) v = x[last + d];
  else if (d < 2 * Hp) v = adjG[last + (d - Hp)];
  else v = coG[last + (d - 2 * Hp)];
  comb[(size_t)b * (3 * Hp) + d] = v;
}

extern "C" void kernel_launch(void* const* d_in, const int* in_sizes, int n_in,
                              void* d_out, int out_size, void* d_ws, size_t ws_size,
                              hipStream_t stream) {
  (void)in_sizes; (void)n_in; (void)out_size; (void)ws_size;
  const int*   item_seq = (const int*)d_in[0];
  const float* tiv      = (const float*)d_in[1];
  const float* item_emb = (const float*)d_in[2];
  const float* pos_emb  = (const float*)d_in[3];
  const float* adj_tab  = (const float*)d_in[4];
  const float* co_tab   = (const float*)d_in[5];
  const float* ln_g  = (const float*)d_in[6];
  const float* ln_b  = (const float*)d_in[7];
  const float* Wq = (const float*)d_in[8];  const float* bq = (const float*)d_in[9];
  const float* Wk = (const float*)d_in[10]; const float* bk = (const float*)d_in[11];
  const float* Wv = (const float*)d_in[12]; const float* bv = (const float*)d_in[13];
  const float* Wo = (const float*)d_in[14]; const float* bo = (const float*)d_in[15];
  const float* ln1g = (const float*)d_in[16]; const float* ln1b = (const float*)d_in[17];
  const float* W1 = (const float*)d_in[18]; const float* b1 = (const float*)d_in[19];
  const float* W2 = (const float*)d_in[20]; const float* b2 = (const float*)d_in[21];
  const float* ln2g = (const float*)d_in[22]; const float* ln2b = (const float*)d_in[23];
  const float* out_W = (const float*)d_in[24]; const float* out_b = (const float*)d_in[25];

  float* ws = (float*)d_ws;
  float* x    = ws; ws += (size_t)NTOK * Hp;
  float* se   = ws; ws += (size_t)NTOK * Hp;
  float* adjG = ws; ws += (size_t)NTOK * Hp;
  float* coG  = ws; ws += (size_t)NTOK * Hp;
  float* qbuf = ws; ws += (size_t)NTOK * Hp;
  float* kbuf = ws; ws += (size_t)NTOK * Hp;
  float* vbuf = ws; ws += (size_t)NTOK * Hp;
  float* ctx  = ws; ws += (size_t)NTOK * Hp;
  float* t0   = ws; ws += (size_t)NTOK * Hp;
  float* ff   = ws; ws += (size_t)NTOK * INNERp;
  float* sf   = ws; ws += (size_t)NTOK * Hp;
  float* af   = ws; ws += (size_t)NTOK * Hp;
  float* cf   = ws; ws += (size_t)NTOK * Hp;
  float* upos = ws; ws += NTOK;
  float* ipos = ws; ws += NTOK;
  float* conf = ws; ws += Bsz;
  float* gmean = ws; ws += 1;
  float* comb = ws; ws += (size_t)Bsz * 3 * Hp;

  float* pred = (float*)d_out;
  float* scal = pred + (size_t)Bsz * Vp;

  // embeddings + input layernorm
  gather_kernel<<<NTOK, Hp, 0, stream>>>(item_emb, item_seq, nullptr, se);
  gather_kernel<<<NTOK, Hp, 0, stream>>>(item_emb, item_seq, pos_emb, t0);
  ln_kernel<<<NTOK, 32, 0, stream>>>(t0, nullptr, ln_g, ln_b, x);

  // transformer layers
  for (int l = 0; l < NLp; ++l) {
    dim3 ghh(NTOK / 16, Hp / 16);
    gemm_wmma<<<ghh, 32, 0, stream>>>(x, Wq + l * Hp * Hp, bq + l * Hp, qbuf, NTOK, Hp, Hp, 0);
    gemm_wmma<<<ghh, 32, 0, stream>>>(x, Wk + l * Hp * Hp, bk + l * Hp, kbuf, NTOK, Hp, Hp, 0);
    gemm_wmma<<<ghh, 32, 0, stream>>>(x, Wv + l * Hp * Hp, bv + l * Hp, vbuf, NTOK, Hp, Hp, 0);
    attn_kernel<<<Bsz * NHp, 64, 0, stream>>>(qbuf, kbuf, vbuf, item_seq, ctx);
    gemm_wmma<<<ghh, 32, 0, stream>>>(ctx, Wo + l * Hp * Hp, bo + l * Hp, t0, NTOK, Hp, Hp, 0);
    ln_kernel<<<NTOK, 32, 0, stream>>>(t0, x, ln1g + l * Hp, ln1b + l * Hp, x);
    gemm_wmma<<<dim3(NTOK / 16, INNERp / 16), 32, 0, stream>>>(
        x, W1 + l * Hp * INNERp, b1 + l * INNERp, ff, NTOK, INNERp, Hp, 1);
    gemm_wmma<<<ghh, 32, 0, stream>>>(ff, W2 + l * INNERp * Hp, b2 + l * Hp, t0, NTOK, Hp, INNERp, 0);
    ln_kernel<<<NTOK, 32, 0, stream>>>(t0, x, ln2g + l * Hp, ln2b + l * Hp, x);
  }

  // feature gathers, conformity, normalization
  gather_kernel<<<NTOK, Hp, 0, stream>>>(adj_tab, item_seq, nullptr, adjG);
  gather_kernel<<<NTOK, Hp, 0, stream>>>(co_tab, item_seq, nullptr, coG);
  tmean_kernel<<<1, 256, 0, stream>>>(tiv, gmean);
  conf_kernel<<<Bsz, 32, 0, stream>>>(tiv, adjG, se, gmean, conf);
  unit_kernel<<<NTOK, 32, 0, stream>>>(x, sf);
  unit_kernel<<<NTOK, 32, 0, stream>>>(adjG, af);
  unit_kernel<<<NTOK, 32, 0, stream>>>(coG, cf);

  // fused NxN cosine-sim + logsumexp (flash style, no NxN materialization)
  sim_lse_wmma<<<NTOK / 16, 32, 0, stream>>>(sf, af, upos, NTOK);
  sim_lse_wmma<<<NTOK / 16, 32, 0, stream>>>(af, cf, ipos, NTOK);
  loss_kernel<<<1, 128, 0, stream>>>(upos, ipos, conf, scal);

  // prediction head
  combined_kernel<<<Bsz, 3 * Hp, 0, stream>>>(x, adjG, coG, comb);
  gemm_wmma<<<dim3(Bsz / 16, Vp / 16), 32, 0, stream>>>(comb, out_W, out_b, pred,
                                                        Bsz, Vp, 3 * Hp, 0);
}
